// ModConv_12876311953949
// MI455X (gfx1250) — matrix operand order repaired and checked by reading, not compiled, not run on hardware
//
#include <hip/hip_runtime.h>

typedef __attribute__((ext_vector_type(16))) __bf16 v16bf;
typedef __attribute__((ext_vector_type(8)))  float  v8f;

#define C_     512
#define HW     4096
#define CK     4608          // C_ * 9 taps
#define TILE_M 128
#define TILE_N 128
#define LDSROW 80            // 64B data (32 bf16) + 16B pad per row
#define ATILE  (128 * LDSROW)        // 10240 B per tap-tile
#define ASET   (9 * ATILE)           // 92160 B: all 9 taps for one 32-ch chunk
#define HALO_ROWS 264                // 4 y-rows * 66 x-cols
#define XTILE  (HALO_ROWS * LDSROW)  // 21120 B
#define XBASE  ASET
#define LDS_SCALES (ASET + 2 * XTILE)        // 134400

// pack two floats into a packed bf16 pair via HW converter (RNE)
__device__ inline unsigned cvt2_bf16(float f0, float f1) {
    union { __bf16 h[2]; unsigned u; } p;
    p.h[0] = (__bf16)f0;     // low 16 bits = even k
    p.h[1] = (__bf16)f1;
    return p.u;
}

// ---------- kernel 1: s[b,ci] = w[b,:] . affine_w[ci,:] + affine_b[ci] + 1 ----------
__global__ __launch_bounds__(256) void mod_scale_kernel(
    const float* __restrict__ w, const float* __restrict__ aw,
    const float* __restrict__ ab, float* __restrict__ s_out)
{
    int g  = blockIdx.x * 256 + threadIdx.x;     // b*512 + ci
    int b  = g >> 9;
    int ci = g & 511;
    const float* wrow = w  + (size_t)b  * 512;   // uniform per block
    const float* arow = aw + (size_t)ci * 512;
    float acc = 0.f;
#pragma unroll 8
    for (int k = 0; k < 512; ++k) acc += wrow[k] * arow[k];
    s_out[g] = acc + ab[ci] + 1.0f;
}

// ---------- kernel 2: d[b,co] = rsqrt( sum_{ci,tap} (weight*s)^2 + eps ) ----------
__global__ __launch_bounds__(256) void demod_kernel(
    const float* __restrict__ weight, const float* __restrict__ s,
    float* __restrict__ d_out)
{
    int bco = blockIdx.x;                        // b*512 + co
    int b   = bco >> 9;
    int co  = bco & 511;
    const float* wrow = weight + (size_t)co * CK;
    const float* srow = s + (size_t)b * 512;
    float acc = 0.f;
    for (int idx = threadIdx.x; idx < CK; idx += 256) {
        float v = wrow[idx] * srow[idx / 9];     // weight layout [co][ci][tap]
        acc += v * v;
    }
    __shared__ float red[256];
    red[threadIdx.x] = acc;
    __syncthreads();
    for (int off = 128; off > 0; off >>= 1) {
        if (threadIdx.x < off) red[threadIdx.x] += red[threadIdx.x + off];
        __syncthreads();
    }
    if (threadIdx.x == 0) d_out[bco] = rsqrtf(red[0] + 1e-8f);
}

// ---------- kernel 3: implicit-GEMM modulated conv via bf16 WMMA ----------
// Per 32-channel chunk (kc): stage ALL 9 taps' modulated A-tiles with coalesced
// b128 loads (taps are contiguous in [co][ci][tap]), plus one zero-padded x halo
// tile; then run 9 staging-free WMMA steps with compile-time (dy,dx) shifts.
__global__ __launch_bounds__(256) void modconv_wmma_kernel(
    const float* __restrict__ x,       // [b][ci][64][64]
    const float* __restrict__ weight,  // [co][ci][3][3]
    const float* __restrict__ s_all,   // [b][512]
    const float* __restrict__ d_all,   // [b][512]
    float* __restrict__ out)           // [b][co][64][64]
{
    // layout: A-set (9 tap tiles) | X0 | X1 | sS(512 f32) | dS(128 f32)
    __shared__ __align__(16) char lds[LDS_SCALES + 512 * 4 + 128 * 4];
    float* sS = (float*)(lds + LDS_SCALES);
    float* dS = sS + 512;

    const int tid  = threadIdx.x;
    const int lane = tid & 31;
    const int wave = tid >> 5;
    const int wm   = wave >> 2;              // 0..1  (64-row co band)
    const int wn   = wave & 3;               // 0..3  (32-col px band)

    const int px0 = blockIdx.x * TILE_N;     // 32 pixel-blocks (2 image rows each)
    const int y0  = px0 >> 6;                // first image row of this block
    const int co0 = blockIdx.y * TILE_M;     // 4 co-blocks
    const int b   = blockIdx.z;              // 16 samples

    // preload per-block scale tables
    sS[tid]       = s_all[b * 512 + tid];
    sS[tid + 256] = s_all[b * 512 + tid + 256];
    if (tid < 128) dS[tid] = d_all[b * 512 + co0 + tid];
    __syncthreads();

    const float* xb = x + (size_t)b * C_ * HW;

    v8f acc[4][2];
    {
        v8f z = {0.f, 0.f, 0.f, 0.f, 0.f, 0.f, 0.f, 0.f};
#pragma unroll
        for (int i = 0; i < 4; ++i)
#pragma unroll
            for (int j = 0; j < 2; ++j) acc[i][j] = z;
    }

    // staging coordinates: 256 threads = (row, k-half)
    const int srow  = tid >> 1;              // 0..127
    const int khalf = tid & 1;               // which 16 channels of the 32-chunk

    // ---- stage all 9 modulated A tiles for channel chunk kc (coalesced b128) ----
    auto stageA = [&](int kc) {
        const int ci0 = kc * 32 + khalf * 16;
        const int co  = co0 + srow;
        const float dv = dS[srow];
        // 16 ci x 9 taps = 144 CONTIGUOUS floats, 16B aligned
        const float* wsrc = weight + (size_t)co * CK + (size_t)ci0 * 9;
        __builtin_prefetch(wsrc + 32 * 9, 0, 1);          // next chunk
        char* abase = lds + srow * LDSROW + (khalf * 16) * 2;
#pragma unroll
        for (int jj = 0; jj < 4; ++jj) {                  // 4 ci per sub-chunk
            union { float4 q[9]; float f[36]; } W;        // 4 ci x 9 taps
            const float4* wv = (const float4*)(wsrc + jj * 36);
#pragma unroll
            for (int i = 0; i < 9; ++i) W.q[i] = wv[i];
            float sd[4];
#pragma unroll
            for (int c = 0; c < 4; ++c) sd[c] = sS[ci0 + jj * 4 + c] * dv;
#pragma unroll
            for (int t = 0; t < 9; ++t) {
                unsigned p0 = cvt2_bf16(W.f[0 * 9 + t] * sd[0], W.f[1 * 9 + t] * sd[1]);
                unsigned p1 = cvt2_bf16(W.f[2 * 9 + t] * sd[2], W.f[3 * 9 + t] * sd[3]);
                char* dstp = abase + t * ATILE + jj * 8;
                *(unsigned*)(dstp)     = p0;
                *(unsigned*)(dstp + 4) = p1;
            }
        }
    };

    // ---- stage zero-padded x halo tile for chunk kc: [hy*66+hx][k 32] bf16 ----
    auto stageX = [&](int kc) {
        const int ci0 = kc * 32 + khalf * 16;
        char* ldsX = lds + XBASE + (kc & 1) * XTILE;
        for (int r = srow; r < HALO_ROWS; r += 128) {
            const int hy = r / 66;               // 0..3
            const int hx = r - hy * 66;          // 0..65
            const int ys = y0 - 1 + hy;
            const int xs = hx - 1;
            const bool inb = ((unsigned)ys < 64u) && ((unsigned)xs < 64u);
            const float* xsrc = xb + (size_t)ci0 * HW + ys * 64 + xs;
            char* xrow = ldsX + r * LDSROW + khalf * 32;
#pragma unroll
            for (int j = 0; j < 16; j += 2) {
                float f0 = inb ? xsrc[(size_t)j * HW]       : 0.f;
                float f1 = inb ? xsrc[(size_t)(j + 1) * HW] : 0.f;
                *(unsigned*)(xrow + j * 2) = cvt2_bf16(f0, f1);
            }
        }
    };

    // ---- one WMMA step: tap fragment set from A-set + shifted halo reads ----
    auto compute = [&](int kc, int tap) {
        const int dy = tap / 3 - 1, dx = tap % 3 - 1;     // compile-time (unrolled)
        union Frag { v16bf v; uint4 q[2]; };
        Frag a[4], bt[2];

        // A 16x32 bf16: lanes 0-15 -> K 0-7 & 16-23, lanes 16-31 -> K 8-15 & 24-31
        const char* ldsA = lds + tap * ATILE;
        const int aoff = (lane & 16) ? 16 : 0;
#pragma unroll
        for (int mf = 0; mf < 4; ++mf) {
            const char* p = ldsA + (wm * 64 + mf * 16 + (lane & 15)) * LDSROW + aoff;
            a[mf].q[0] = *(const uint4*)(p);
            a[mf].q[1] = *(const uint4*)(p + 32);
        }

        // B 32x16 bf16: lanes 0-15 -> K 0-7 & 16-23, lanes 16-31 -> K 8-15 & 24-31, N = lane%16
        const char* ldsX = lds + XBASE + (kc & 1) * XTILE;
        const int boff = (lane & 16) ? 16 : 0;
#pragma unroll
        for (int nf = 0; nf < 2; ++nf) {
            const int p  = wn * 32 + nf * 16 + (lane & 15);  // 0..127 local pixel
            const int hy = (p >> 6) + dy + 1;                // 0..3
            const int hx = (p & 63) + dx + 1;                // 0..65
            const char* q = ldsX + (hy * 66 + hx) * LDSROW + boff;
            bt[nf].q[0] = *(const uint4*)(q);
            bt[nf].q[1] = *(const uint4*)(q + 32);
        }

#pragma unroll
        for (int mf = 0; mf < 4; ++mf)
#pragma unroll
            for (int nf = 0; nf < 2; ++nf)
                acc[mf][nf] = __builtin_amdgcn_wmma_f32_16x16x32_bf16(
                    false, a[mf].v, false, bt[nf].v,
                    (short)0, acc[mf][nf], false, false);
    };

    // ---- main loop over 16 channel chunks; 72 WMMAs between A restages ----
    stageX(0);
    stageA(0);
    __syncthreads();
    for (int kc = 0; kc < 16; ++kc) {
        if (kc + 1 < 16) stageX(kc + 1);      // overlaps with compute (other X buffer)
#pragma unroll
        for (int tap = 0; tap < 9; ++tap) compute(kc, tap);
        __syncthreads();                      // A-set reads done
        if (kc + 1 < 16) stageA(kc + 1);      // single-buffered A-set restage
        __syncthreads();                      // A-set writes visible
    }

    // ---- epilogue: C layout — VGPR v: M = v (lanes 0-15) / v+8 (lanes 16-31), N = lane%16 ----
    const int co_base = co0 + wm * 64;
    const int px_base = px0 + wn * 32;
    const int nlane   = lane & 15;
    const int mhi     = (lane >> 4) * 8;
#pragma unroll
    for (int mf = 0; mf < 4; ++mf)
#pragma unroll
        for (int nf = 0; nf < 2; ++nf)
#pragma unroll
            for (int v = 0; v < 8; ++v) {
                int co = co_base + mf * 16 + mhi + v;
                int px = px_base + nf * 16 + nlane;
                out[((size_t)b * C_ + co) * HW + px] = acc[mf][nf][v];
            }
}

// ---------- launcher ----------
extern "C" void kernel_launch(void* const* d_in, const int* in_sizes, int n_in,
                              void* d_out, int out_size, void* d_ws, size_t ws_size,
                              hipStream_t stream) {
    const float* x  = (const float*)d_in[0];   // [16,512,64,64]
    const float* w  = (const float*)d_in[1];   // [16,512]
    const float* wt = (const float*)d_in[2];   // [512,512,3,3]
    const float* aw = (const float*)d_in[3];   // [512,512]
    const float* ab = (const float*)d_in[4];   // [512]
    float* out = (float*)d_out;

    float* s_buf = (float*)d_ws;               // 16*512 floats
    float* d_buf = s_buf + 16 * 512;           // 16*512 floats  (64 KB total)

    mod_scale_kernel<<<32, 256, 0, stream>>>(w, aw, ab, s_buf);
    demod_kernel<<<16 * 512, 256, 0, stream>>>(wt, s_buf, d_buf);

    dim3 grid(32, 4, 16);                      // px-blocks, co-blocks, batch
    modconv_wmma_kernel<<<grid, 256, 0, stream>>>(x, wt, s_buf, d_buf, out);
}
